// LSTMFirstNN_64192581206092
// MI455X (gfx1250) — compile-verified, matrix operand order checked by reading, not run on hardware
//
#include <hip/hip_runtime.h>
#include <hip/hip_bf16.h>

// ---------------------------------------------------------------------------
// Types for CDNA5 WMMA (wave32): bf16 A/B fragments (16 x bf16 = 8 VGPRs),
// f32 accumulator (8 x f32 = 8 VGPRs).
// ---------------------------------------------------------------------------
typedef __attribute__((ext_vector_type(16))) __bf16 v16bf;
typedef __attribute__((ext_vector_type(8)))  __bf16 v8bf;
typedef __attribute__((ext_vector_type(8)))  float  v8f;

struct FragBF {
    union { v16bf v; v8bf h[2]; };
};

// Fast transcendental path: v_exp_f32 / v_rcp_f32, no IEEE-division fixups.
__device__ __forceinline__ float fexp(float x) {
    return __builtin_amdgcn_exp2f(x * 1.4426950408889634f);
}
__device__ __forceinline__ float frcp(float x) {
    return __builtin_amdgcn_rcpf(x);
}
__device__ __forceinline__ float sigf(float x) {
    return frcp(1.0f + fexp(-x));
}
__device__ __forceinline__ float tanh_fast(float x) {
    float t = fminf(fmaxf(x, -15.0f), 15.0f);
    float e = fexp(2.0f * t);
    return (e - 1.0f) * frcp(e + 1.0f);
}

// ---------------------------------------------------------------------------
// Prep kernels: f32 -> bf16 weight conversion, and x [B,F,T] -> xb [B,T,F] bf16
// ---------------------------------------------------------------------------
__global__ void f32_to_bf16_kernel(const float* __restrict__ in,
                                   __bf16* __restrict__ out, int n) {
    int i = blockIdx.x * blockDim.x + threadIdx.x;
    if (i < n) out[i] = (__bf16)in[i];
}

__global__ void transpose_x_kernel(const float* __restrict__ x,  // [B,F,T]
                                   __bf16* __restrict__ xb,      // [B,T,F]
                                   int B, int F, int T) {
    int i = blockIdx.x * blockDim.x + threadIdx.x;
    int total = B * T * F;
    if (i >= total) return;
    int f = i % F;
    int t = (i / F) % T;
    int b = i / (F * T);
    xb[i] = (__bf16)x[((size_t)b * F + f) * T + t];
}

// ---------------------------------------------------------------------------
// Fused LSTM layer (persistent over T inside the block).
//   Block: 256 threads = 8 waves; batch tile M = 16 (one m-tile) so the grid
//   is B/16 = 256 blocks (fills the WGPs; each block carries the serial chain).
//   Wave w owns h-slice w*16, replicated over the 4 gates => 4 WMMA
//   accumulator tiles per wave (i/f/g/o, one 16-col tile each).
//   Whh B-fragments hoisted into 128 VGPRs (loop-invariant over t) so the
//   sequential recurrence is pure back-to-back WMMA; Wih streams from LDS
//   with batched loads. Bias folded into the activation epilogue.
//   ATTN=true folds logits + online softmax + attended/final (flash-style),
//   with parity-double-buffered logit scratch => 2 barriers per step.
// ---------------------------------------------------------------------------
template <int KIN, bool ATTN>
__global__ __launch_bounds__(256) void lstm_kernel(
    const __bf16* __restrict__ xin,   // [B,T,KIN] bf16
    const __bf16* __restrict__ wihg,  // [512,KIN] bf16
    const __bf16* __restrict__ whhg,  // [512,128] bf16
    const float*  __restrict__ bih,   // [512]
    const float*  __restrict__ bhh,   // [512]
    const float*  __restrict__ wa,    // [128]   (ATTN only)
    const float*  __restrict__ ba,    // [1]     (ATTN only)
    __bf16*       __restrict__ hout,  // [B,T,128] bf16 (!ATTN only)
    float*        __restrict__ attended, // [B,128] f32 (ATTN only)
    float*        __restrict__ finalh)   // [B,128] f32 (ATTN only)
{
    constexpr int T = 150, H = 128, NG = 512, MTILE = 16;

    extern __shared__ char smem[];
    __bf16* s_wih  = (__bf16*)smem;              // NG*KIN
    __bf16* s_whh  = s_wih + NG * KIN;           // NG*H
    __bf16* s_ht   = s_whh + NG * H;             // MTILE*H
    float*  s_bias = (float*)(s_ht + MTILE * H); // NG
    float*  s_wa   = s_bias + NG;                // H
    float*  s_lg   = s_wa + H;                   // 2*MTILE (parity buffers)

    const int tid  = threadIdx.x;
    const int lane = tid & 31;
    const int wv   = tid >> 5;
    const int hs   = wv * 16;          // h-slice base (8 slices of 16)
    const int ln16 = lane & 15;
    const bool hi  = lane >= 16;
    const int b0   = blockIdx.x * MTILE;

    // ---- cooperative LDS fill ----
    {
        const int4* s1 = (const int4*)wihg; int4* d1 = (int4*)s_wih;
        const int n1 = NG * KIN * 2 / 16;
        for (int i = tid; i < n1; i += 256) d1[i] = s1[i];
        const int4* s2 = (const int4*)whhg; int4* d2 = (int4*)s_whh;
        const int n2 = NG * H * 2 / 16;
        for (int i = tid; i < n2; i += 256) d2[i] = s2[i];
        for (int i = tid; i < NG; i += 256) s_bias[i] = bih[i] + bhh[i];
        if (ATTN) for (int i = tid; i < H; i += 256) s_wa[i] = wa[i];
        int4 z = {0, 0, 0, 0};
        int4* dz = (int4*)s_ht;
        for (int i = tid; i < MTILE * H * 2 / 16; i += 256) dz[i] = z;
        if (tid < 2 * MTILE) s_lg[tid] = 0.0f;
    }
    __syncthreads();

    const int kofA = hi ? 8 : 0;   // 16-bit fragment layout: lanes>=16 hold K+8/K+24

    // ---- hoist Whh B-fragments (loop-invariant over t): 16 frags = 128 VGPRs
    FragBF wB[4][4];
#pragma unroll
    for (int ks = 0; ks < 4; ++ks)
#pragma unroll
        for (int g = 0; g < 4; ++g) {
            int n = g * H + hs + ln16;
            const __bf16* bp = s_whh + n * H + ks * 32 + kofA;
            wB[ks][g].h[0] = *(const v8bf*)(bp);
            wB[ks][g].h[1] = *(const v8bf*)(bp + 16);
        }

    // ---- hoist per-tile bias + attention weight (loop-invariant scalars) ---
    float bg[4];
#pragma unroll
    for (int g = 0; g < 4; ++g)
        bg[g] = s_bias[g * H + hs + ln16];
    const float waL = ATTN ? s_wa[hs + ln16] : 0.0f;
    const float bav = ATTN ? ba[0] : 0.0f;

    float cst[8], hv[8];
#pragma unroll
    for (int r = 0; r < 8; ++r) { cst[r] = 0.0f; hv[r] = 0.0f; }

    float mrun[8], lrun[8], av[8];
    if (ATTN) {
#pragma unroll
        for (int r = 0; r < 8; ++r) {
            mrun[r] = -3.0e38f; lrun[r] = 0.0f; av[r] = 0.0f;
        }
    }

    // Batch row this lane supplies for A fragments (A: M = lane%16).
    const __bf16* xrow = xin + (size_t)(b0 + ln16) * T * KIN;

    const v8f zc = {};

    for (int t = 0; t < T; ++t) {
        v8f acc2[4];

        // ---- input projection: gates = x_t[16xKIN] * Wih^T  (zero-C start) --
#pragma unroll
        for (int ks = 0; ks < KIN / 32; ++ks) {
            FragBF A;
            const __bf16* ap = xrow + (size_t)t * KIN + ks * 32 + kofA;
            A.h[0] = *(const v8bf*)(ap);
            A.h[1] = *(const v8bf*)(ap + 16);
            FragBF Bf[4];
#pragma unroll
            for (int g = 0; g < 4; ++g) {
                int n = g * H + hs + ln16;
                const __bf16* bp = s_wih + n * KIN + ks * 32 + kofA;
                Bf[g].h[0] = *(const v8bf*)(bp);
                Bf[g].h[1] = *(const v8bf*)(bp + 16);
            }
#pragma unroll
            for (int g = 0; g < 4; ++g)
                acc2[g] = __builtin_amdgcn_wmma_f32_16x16x32_bf16(
                    false, A.v, false, Bf[g].v, (short)0,
                    (ks == 0) ? zc : acc2[g], false, false);
        }

        // prefetch next timestep's x row (global_prefetch)
        if (t + 1 < T) {
            const char* pf = (const char*)(xrow + (size_t)(t + 1) * KIN);
            __builtin_prefetch(pf, 0, 3);
            __builtin_prefetch(pf + 128, 0, 3);
        }

        // ---- recurrence: gates += h[16x128] * Whh^T (register-resident B) --
#pragma unroll
        for (int ks = 0; ks < 4; ++ks) {
            FragBF A;
            const __bf16* ap = s_ht + ln16 * H + ks * 32 + kofA;
            A.h[0] = *(const v8bf*)(ap);
            A.h[1] = *(const v8bf*)(ap + 16);
#pragma unroll
            for (int g = 0; g < 4; ++g)
                acc2[g] = __builtin_amdgcn_wmma_f32_16x16x32_bf16(
                    false, A.v, false, wB[ks][g].v, (short)0, acc2[g], false, false);
        }

        __syncthreads();   // all reads of s_ht (and prev-parity s_lg) complete

        float* lg_cur = s_lg + (t & 1) * MTILE;
        float* lg_nxt = s_lg + ((t + 1) & 1) * MTILE;

        // ---- LSTM cell update + write new h ----
#pragma unroll
        for (int r = 0; r < 8; ++r) {
            float iv = sigf(acc2[0][r] + bg[0]);
            float fv = sigf(acc2[1][r] + bg[1]);
            float gv = tanh_fast(acc2[2][r] + bg[2]);
            float ov = sigf(acc2[3][r] + bg[3]);
            float c  = fv * cst[r] + iv * gv;
            cst[r] = c;
            float h  = ov * tanh_fast(c);
            hv[r] = h;
            int mrow = hi ? 8 + r : r;          // C layout: M=r / M=8+r
            s_ht[mrow * H + hs + ln16] = (__bf16)h;
        }

        if (ATTN) {
            // zero next-parity logit buffer (no extra barrier needed)
            if (tid < MTILE) lg_nxt[tid] = 0.0f;
            // ---- attention logits: dot(h_b, Wa) reduced across the block ---
#pragma unroll
            for (int r = 0; r < 8; ++r) {
                float p = hv[r] * waL;
#pragma unroll
                for (int off = 1; off < 16; off <<= 1)
                    p += __shfl_xor(p, off, 32);
                if (ln16 == 0)
                    atomicAdd(&lg_cur[hi ? 8 + r : r], p);
            }
        }
        __syncthreads();                 // new h + logits visible

        if (ATTN) {
            // ---- online softmax accumulation (flash-style) ----
#pragma unroll
            for (int r = 0; r < 8; ++r) {
                float s  = lg_cur[hi ? 8 + r : r] + bav;
                float mn = fmaxf(mrun[r], s);
                float sc = fexp(mrun[r] - mn);
                float w  = fexp(s - mn);
                lrun[r]  = lrun[r] * sc + w;
                av[r]    = av[r] * sc + w * hv[r];
                mrun[r]  = mn;
            }
        } else {
            // ---- coalesced b128 export of the new h tile to global ----
            int mrow = tid >> 4;             // 16 int4 per 128-col bf16 row
            int c16  = tid & 15;
            int4 val = ((const int4*)s_ht)[tid];
            int4* dst = (int4*)((char*)hout +
                        (((size_t)(b0 + mrow) * T + t) * (H * 2)));
            dst[c16] = val;
        }
    }

    if (ATTN) {
#pragma unroll
        for (int r = 0; r < 8; ++r) {
            int b = b0 + (hi ? 8 + r : r);
            float inv = frcp(lrun[r]);
            attended[(size_t)b * H + hs + ln16] = av[r] * inv;
            finalh [(size_t)b * H + hs + ln16] = hv[r];   // h at t = T-1
        }
    }
}

// ---------------------------------------------------------------------------
// Head: y1=relu(fin W1^T+b1)*bn1 ; y2=relu(y1 W2^T+b2)*bn2 ;
//       comb=[y2, attended] ; z=relu(comb Wc1^T+bc1) ; out=z Wc2^T+bc2
// One block (256 thr) per batch row; tiny GEMVs via LDS staging.
// ---------------------------------------------------------------------------
__global__ __launch_bounds__(256) void head_kernel(
    const float* __restrict__ fin, const float* __restrict__ att,
    const float* __restrict__ W1, const float* __restrict__ b1,
    const float* __restrict__ g1, const float* __restrict__ be1,
    const float* __restrict__ W2, const float* __restrict__ b2,
    const float* __restrict__ g2, const float* __restrict__ be2,
    const float* __restrict__ Wc1, const float* __restrict__ bc1,
    const float* __restrict__ Wc2, const float* __restrict__ bc2,
    float* __restrict__ out)
{
    __shared__ float sf[128], y1[256], comb[256], z1[128];
    const float rsq = rsqrtf(1.0f + 1e-5f);
    int b = blockIdx.x, tid = threadIdx.x;

    if (tid < 128) {
        sf[tid]         = fin[(size_t)b * 128 + tid];
        comb[128 + tid] = att[(size_t)b * 128 + tid];
    }
    __syncthreads();
    {
        float acc = b1[tid];
        const float* w = &W1[tid * 128];
        for (int k = 0; k < 128; ++k) acc += sf[k] * w[k];
        acc = fmaxf(acc, 0.0f);
        y1[tid] = acc * (g1[tid] * rsq) + be1[tid];
    }
    __syncthreads();
    if (tid < 128) {
        float acc = b2[tid];
        const float* w = &W2[tid * 256];
        for (int k = 0; k < 256; ++k) acc += y1[k] * w[k];
        acc = fmaxf(acc, 0.0f);
        comb[tid] = acc * (g2[tid] * rsq) + be2[tid];
    }
    __syncthreads();
    if (tid < 128) {
        float acc = bc1[tid];
        const float* w = &Wc1[tid * 256];
        for (int k = 0; k < 256; ++k) acc += comb[k] * w[k];
        z1[tid] = fmaxf(acc, 0.0f);
    }
    __syncthreads();
    if (tid < 4) {
        float acc = bc2[tid];
        const float* w = &Wc2[tid * 128];
        for (int k = 0; k < 128; ++k) acc += z1[k] * w[k];
        out[(size_t)b * 4 + tid] = acc;
    }
}

// ---------------------------------------------------------------------------
extern "C" void kernel_launch(void* const* d_in, const int* in_sizes, int n_in,
                              void* d_out, int out_size, void* d_ws, size_t ws_size,
                              hipStream_t stream)
{
    constexpr int B = 4096, F = 64, T = 150, H = 128, NG = 512;

    const float* x    = (const float*)d_in[0];
    const float* Wih0 = (const float*)d_in[1];
    const float* Whh0 = (const float*)d_in[2];
    const float* bih0 = (const float*)d_in[3];
    const float* bhh0 = (const float*)d_in[4];
    const float* Wih1 = (const float*)d_in[5];
    const float* Whh1 = (const float*)d_in[6];
    const float* bih1 = (const float*)d_in[7];
    const float* bhh1 = (const float*)d_in[8];
    const float* Wa   = (const float*)d_in[9];
    const float* ba   = (const float*)d_in[10];
    const float* W1   = (const float*)d_in[11];
    const float* b1   = (const float*)d_in[12];
    const float* g1   = (const float*)d_in[13];
    const float* be1  = (const float*)d_in[14];
    const float* W2   = (const float*)d_in[15];
    const float* b2   = (const float*)d_in[16];
    const float* g2   = (const float*)d_in[17];
    const float* be2  = (const float*)d_in[18];
    const float* Wc1  = (const float*)d_in[19];
    const float* bc1  = (const float*)d_in[20];
    const float* Wc2  = (const float*)d_in[21];
    const float* bc2  = (const float*)d_in[22];
    float* out = (float*)d_out;

    char* ws = (char*)d_ws;
    size_t off = 0;
    auto walloc = [&](size_t bytes) -> char* {
        char* p = ws + off;
        off += (bytes + 255) & ~(size_t)255;
        return p;
    };
    __bf16* xb    = (__bf16*)walloc((size_t)B * T * F * 2);   // transposed bf16 input
    __bf16* h1    = (__bf16*)walloc((size_t)B * T * H * 2);   // layer-0 output
    __bf16* wih0b = (__bf16*)walloc((size_t)NG * F * 2);
    __bf16* whh0b = (__bf16*)walloc((size_t)NG * H * 2);
    __bf16* wih1b = (__bf16*)walloc((size_t)NG * H * 2);
    __bf16* whh1b = (__bf16*)walloc((size_t)NG * H * 2);
    float*  attn  = (float*)walloc((size_t)B * H * 4);
    float*  finh  = (float*)walloc((size_t)B * H * 4);

    // --- prep: weight conversion + input transpose ---
    f32_to_bf16_kernel<<<(NG * F + 255) / 256, 256, 0, stream>>>(Wih0, wih0b, NG * F);
    f32_to_bf16_kernel<<<(NG * H + 255) / 256, 256, 0, stream>>>(Whh0, whh0b, NG * H);
    f32_to_bf16_kernel<<<(NG * H + 255) / 256, 256, 0, stream>>>(Wih1, wih1b, NG * H);
    f32_to_bf16_kernel<<<(NG * H + 255) / 256, 256, 0, stream>>>(Whh1, whh1b, NG * H);
    transpose_x_kernel<<<((size_t)B * T * F + 255) / 256, 256, 0, stream>>>(x, xb, B, F, T);

    // --- dynamic LDS: weights + bias + h-tile + attn scratch ---
    auto shbytes = [](int kin) -> size_t {
        return (size_t)(NG * kin + NG * H + 16 * H) * 2 + (size_t)(NG + H + 32) * 4;
    };

    lstm_kernel<F, false><<<B / 16, 256, shbytes(F), stream>>>(
        xb, wih0b, whh0b, bih0, bhh0, nullptr, nullptr, h1, nullptr, nullptr);

    lstm_kernel<H, true><<<B / 16, 256, shbytes(H), stream>>>(
        h1, wih1b, whh1b, bih1, bhh1, Wa, ba, nullptr, attn, finh);

    head_kernel<<<B, 256, 0, stream>>>(finh, attn,
        W1, b1, g1, be1, W2, b2, g2, be2, Wc1, bc1, Wc2, bc2, out);
}
